// GNN_Encoder_34067680592318
// MI455X (gfx1250) — compile-verified, matrix operand order checked by reading, not run on hardware
//
#include <hip/hip_runtime.h>
#include <hip/hip_bf16.h>

typedef __attribute__((ext_vector_type(16))) _Float16     v16h;
typedef __attribute__((ext_vector_type(8)))  float        v8f;
typedef __attribute__((ext_vector_type(4)))  unsigned int uint4v;

union FragU { uint4v q[2]; v16h h; };

#define NWAVES 8
#define GROW   72   // halves per row in per-wave relayout buffer (bank-spread pad)

struct __align__(16) Smem {
  unsigned int w1sw[4 * 256];     //  4 KB : W1  B-fragments (K 4 zero-padded to 32), 4 n-tiles
  unsigned int w2sw[8 * 256];     //  8 KB : W2  B-fragments, (nt, kf)
  unsigned int wfcsw[32 * 256];   // 32 KB : Wfc B-fragments, (nt, kf)
  float b1[64];
  float b2[64];
  float bfc[256];
  _Float16 gbuf[NWAVES][16 * GROW]; // per-wave D->A relayout buffer (f16, row-major)
};

static __device__ inline unsigned pack2h(float lo, float hi) {
  union { _Float16 h[2]; unsigned u; } t;
  t.h[0] = (_Float16)lo; t.h[1] = (_Float16)hi;
  return t.u;
}

static __device__ inline float lrelu(float x) { return fmaxf(x, 0.01f * x); }

static __device__ inline v8f wmma16(v16h a, v16h b, v8f c) {
  return __builtin_amdgcn_wmma_f32_16x16x32_f16(false, a, false, b, (short)0, c,
                                                false, false);
}

// B fragment: swizzled weights, lane-major 32B per lane -> two ds_load_b128
static __device__ inline v16h ld_bfrag(const unsigned int* sw, int frag, int lane) {
  FragU u;
  const uint4v* p = (const uint4v*)(sw + frag * 256 + lane * 8);
  u.q[0] = p[0];
  u.q[1] = p[1];
  return u.h;
}

// A fragment from row-major f16 [16][GROW] buffer.
// lane<16: K 0..7 @ +0B and K 16..23 @ +32B ; lane>=16: +16B / +48B ; kf adds 64B
static __device__ inline v16h ld_afrag(const _Float16* buf, int row, int hiOff, int kf) {
  FragU u;
  const uint4v* p = (const uint4v*)(buf + row * GROW + hiOff + kf * 32);
  u.q[0] = p[0];
  u.q[1] = p[2];
  return u.h;
}

__global__ __launch_bounds__(256)
void gnn_encoder_kernel(const float* __restrict__ state,
                        const float* __restrict__ W1, const float* __restrict__ b1,
                        const float* __restrict__ W2, const float* __restrict__ b2,
                        const float* __restrict__ Wfc, const float* __restrict__ bfc,
                        float* __restrict__ out, int nElems, int nTiles) {
  __shared__ Smem sm;
  const int tid  = threadIdx.x;
  const int lane = tid & 31;
  const int wave = tid >> 5;

  // ------- one-time: convert weights f32->f16 and swizzle into B-fragment layout -------
  for (int d = tid; d < 4 * 256; d += 256) {          // W1 (K padded 4->32)
    int f = d >> 8, s = d & 255, ln = s >> 3, v = s & 7;
    int n = f * 16 + (ln & 15);
    int k = ((v >= 4) ? 16 : 0) + ((ln >= 16) ? 8 : 0) + (v & 3) * 2;
    float lo = (k     < 4) ? W1[k * 64 + n]       : 0.f;
    float hi = (k + 1 < 4) ? W1[(k + 1) * 64 + n] : 0.f;
    sm.w1sw[d] = pack2h(lo, hi);
  }
  for (int d = tid; d < 8 * 256; d += 256) {          // W2: frag = nt*2 + kf
    int f = d >> 8, kf = f & 1, nt = f >> 1;
    int s = d & 255, ln = s >> 3, v = s & 7;
    int n = nt * 16 + (ln & 15);
    int k = kf * 32 + ((v >= 4) ? 16 : 0) + ((ln >= 16) ? 8 : 0) + (v & 3) * 2;
    sm.w2sw[d] = pack2h(W2[k * 64 + n], W2[(k + 1) * 64 + n]);
  }
  for (int d = tid; d < 32 * 256; d += 256) {         // Wfc: frag = nt*2 + kf
    int f = d >> 8, kf = f & 1, nt = f >> 1;
    int s = d & 255, ln = s >> 3, v = s & 7;
    int n = nt * 16 + (ln & 15);
    int k = kf * 32 + ((v >= 4) ? 16 : 0) + ((ln >= 16) ? 8 : 0) + (v & 3) * 2;
    sm.wfcsw[d] = pack2h(Wfc[k * 256 + n], Wfc[(k + 1) * 256 + n]);
  }
  if (tid < 64) { sm.b1[tid] = b1[tid]; sm.b2[tid] = b2[tid]; }
  sm.bfc[tid] = bfc[tid];
  __syncthreads();

  const float C1 = 0.35355339059327373f;   // A_hat off-diag (1/(2*sqrt(2))*2 form)
  const int r      = lane & 15;            // A-fragment row this lane owns
  const int nodeId = r & 3;
  const int eOff   = r >> 2;
  const int nCol   = lane & 15;            // D-fragment column this lane owns
  const int rbase  = (lane < 16) ? 0 : 8;  // first D row in this lane's VGPRs
  const int hiOff  = (lane >= 16) ? 8 : 0; // A-fragment K-half offset (halves)
  _Float16* gb = sm.gbuf[wave];

  for (int t = blockIdx.x; t < nTiles; t += gridDim.x) {
    const long e0 = (long)t * 32 + wave * 4;   // first element of this wave's 16-row tile

    // ---- layer-1 A fragment built straight from state (A_hat folded in) ----
    v16h a1;
    #pragma unroll
    for (int i = 0; i < 16; ++i) a1[i] = (_Float16)0.f;
    {
      long e = e0 + eOff;
      if (e >= nElems) e = nElems - 1;
      const float* st = state + e * 12;
      if (lane < 16) {                      // only K=0..3 nonzero (lanes 0-15, VGPR0-1)
        float x0, x1, x2, x3;
        if (nodeId == 0) {
          x0 = C1 * (st[0] + st[4] + st[8]);
          x1 = C1 * (st[1] + st[5] + st[9]);
          x2 = C1 * (st[2] + st[6] + st[10]);
          x3 = C1 * (st[3] + st[7] + st[11]);
        } else {
          const float* sn = st + (nodeId - 1) * 4;
          x0 = 0.5f * sn[0]; x1 = 0.5f * sn[1]; x2 = 0.5f * sn[2]; x3 = 0.5f * sn[3];
        }
        a1[0] = (_Float16)x0; a1[1] = (_Float16)x1;
        a1[2] = (_Float16)x2; a1[3] = (_Float16)x3;
      }
    }

    // ---- layer 1: WMMA + bias + lrelu + register-local A_hat mix -> g in LDS ----
    #pragma unroll
    for (int nt = 0; nt < 4; ++nt) {
      v8f acc;
      #pragma unroll
      for (int i = 0; i < 8; ++i) acc[i] = 0.f;
      acc = wmma16(a1, ld_bfrag(sm.w1sw, nt, lane), acc);
      const float bias = sm.b1[nt * 16 + nCol];
      float h[8], g[8];
      #pragma unroll
      for (int i = 0; i < 8; ++i) h[i] = lrelu(acc[i] + bias);
      #pragma unroll
      for (int q = 0; q < 2; ++q) {         // two elements live in this lane's 8 rows
        const float* hh = h + q * 4;
        float*       gg = g + q * 4;
        float s = hh[1] + hh[2] + hh[3];
        gg[0] = 0.25f * hh[0] + C1 * s;
        gg[1] = C1 * hh[0] + 0.5f * hh[1];
        gg[2] = C1 * hh[0] + 0.5f * hh[2];
        gg[3] = C1 * hh[0] + 0.5f * hh[3];
      }
      const int col = nt * 16 + nCol;
      #pragma unroll
      for (int j = 0; j < 8; ++j) gb[(rbase + j) * GROW + col] = (_Float16)g[j];
    }

    // ---- layer 2: [16x64] @ W2[64x64] ----
    v16h a20 = ld_afrag(gb, r, hiOff, 0);
    v16h a21 = ld_afrag(gb, r, hiOff, 1);
    #pragma unroll
    for (int nt = 0; nt < 4; ++nt) {
      v8f acc;
      #pragma unroll
      for (int i = 0; i < 8; ++i) acc[i] = 0.f;
      acc = wmma16(a20, ld_bfrag(sm.w2sw, nt * 2 + 0, lane), acc);
      acc = wmma16(a21, ld_bfrag(sm.w2sw, nt * 2 + 1, lane), acc);
      const float bias = sm.b2[nt * 16 + nCol];
      const int col = nt * 16 + nCol;
      #pragma unroll
      for (int j = 0; j < 8; ++j)
        gb[(rbase + j) * GROW + col] = (_Float16)lrelu(acc[j] + bias);
    }

    // ---- layer 3: [16x64] @ Wfc[64x256], lrelu, mean over 4 node rows, store ----
    v16h a30 = ld_afrag(gb, r, hiOff, 0);
    v16h a31 = ld_afrag(gb, r, hiOff, 1);
    const long eA = e0 + ((lane < 16) ? 0 : 2);
    const long eB = eA + 1;
    #pragma unroll
    for (int nt = 0; nt < 16; ++nt) {
      v8f acc;
      #pragma unroll
      for (int i = 0; i < 8; ++i) acc[i] = 0.f;
      acc = wmma16(a30, ld_bfrag(sm.wfcsw, nt * 2 + 0, lane), acc);
      acc = wmma16(a31, ld_bfrag(sm.wfcsw, nt * 2 + 1, lane), acc);
      const int col = nt * 16 + nCol;
      const float bias = sm.bfc[col];
      float h[8];
      #pragma unroll
      for (int i = 0; i < 8; ++i) h[i] = lrelu(acc[i] + bias);
      const float o0 = 0.25f * (h[0] + h[1] + h[2] + h[3]);
      const float o1 = 0.25f * (h[4] + h[5] + h[6] + h[7]);
      if (eA < nElems) out[eA * 256 + col] = o0;
      if (eB < nElems) out[eB * 256 + col] = o1;
    }
  }
}

extern "C" void kernel_launch(void* const* d_in, const int* in_sizes, int n_in,
                              void* d_out, int out_size, void* d_ws, size_t ws_size,
                              hipStream_t stream) {
  (void)n_in; (void)out_size; (void)d_ws; (void)ws_size;
  const float* state = (const float*)d_in[0];
  const float* W1    = (const float*)d_in[1];
  const float* b1    = (const float*)d_in[2];
  const float* W2    = (const float*)d_in[3];
  const float* b2    = (const float*)d_in[4];
  const float* Wfc   = (const float*)d_in[5];
  const float* bfc   = (const float*)d_in[6];
  float* out = (float*)d_out;

  int nElems = in_sizes[0] / 12;             // [B, 12] fp32 state
  int nTiles = (nElems + 31) / 32;           // 32 elements (8 waves x 4) per block
  int grid   = nTiles < 2048 ? nTiles : 2048;
  if (grid < 1) grid = 1;
  gnn_encoder_kernel<<<grid, 256, 0, stream>>>(state, W1, b1, W2, b2, Wfc, bfc,
                                               out, nElems, nTiles);
}
